// DeformableConv2d_91061896610543
// MI455X (gfx1250) — matrix operand order repaired and checked by reading, not compiled, not run on hardware
//
#include <hip/hip_runtime.h>

typedef __attribute__((ext_vector_type(4)))  __bf16 v4bf;
typedef __attribute__((ext_vector_type(8)))  __bf16 v8bf;
typedef __attribute__((ext_vector_type(16))) __bf16 v16bf;
typedef __attribute__((ext_vector_type(8)))  float  v8f;

#define KN   9
#define CH   128
#define FF   128
#define HH   64
#define WW   64
#define KTOT (KN * CH)          // 1152
#define KSTEPS (KTOT / 32)      // 36
#define A_STRIDE (KTOT + 8)     // 1160 bf16 rows: 16B-aligned, conflict-free b128 reads
#define NPAIR (16 * KN)         // 144 (pixel, tap) pairs per block

// ---------------------------------------------------------------------------
// Pre-pass: W (KN,C,F) f32 row-major  ->  Wt[f][k] bf16, k = n*C + c
// ---------------------------------------------------------------------------
__global__ __launch_bounds__(256)
void weight_prepass_kernel(const float* __restrict__ wgt, __bf16* __restrict__ wt)
{
    const int idx = blockIdx.x * 256 + threadIdx.x;   // over KTOT*FF elements
    if (idx < KTOT * FF) {
        const int k = idx / FF;       // row of W (n*C + c)
        const int f = idx % FF;       // coalesced read along f
        wt[(size_t)f * KTOT + k] = (__bf16)wgt[idx];
    }
}

// ---------------------------------------------------------------------------
// Main fused kernel: bilinear gather -> LDS bf16 panel -> WMMA GEMM
// ---------------------------------------------------------------------------
template <bool BF16W>
__global__ __launch_bounds__(128)
void deform_conv_wmma_kernel(const float*  __restrict__ xin,
                             const float*  __restrict__ off,
                             const float*  __restrict__ wgt,
                             const __bf16* __restrict__ wt,
                             const float*  __restrict__ bias,
                             float*        __restrict__ out)
{
    __shared__ __attribute__((aligned(16))) __bf16 sA[16 * A_STRIDE];
    __shared__ __attribute__((aligned(16))) int4   sOfs[NPAIR];   // lt, rt, lb, rb base idx
    __shared__ __attribute__((aligned(16))) float2 sFw[NPAIR];    // fy, fx
    __shared__ __attribute__((aligned(16))) int    sDst[NPAIR];   // A-panel dest offset

    const int t    = threadIdx.x;        // 0..127
    const int lane = t & 31;
    const int wave = t >> 5;             // 0..3
    const int pbase = blockIdx.x * 16;   // first pixel of this M-tile

    // Faithful TF tap order from stack(meshgrid(ij)).reshape(-1,2)
    const int TAP_I[KN] = {0, 0, 1, 2, 2, 1, 0, 2, 1};
    const int TAP_J[KN] = {0, 1, 1, 2, 0, 2, 1, 0, 2};

    // ---- Stage 0: compute the 144 sample coordinates once (no redundancy)
    for (int idx = t; idx < NPAIR; idx += 128) {
        const int p = idx / KN;
        const int n = idx % KN;
        const int m = pbase + p;
        const int b = m >> 12;               // / (64*64)
        const int h = (m >> 6) & 63;
        const int w = m & 63;
        const float oy = off[(size_t)m * (2 * KN) + 2 * n + 0];
        const float ox = off[(size_t)m * (2 * KN) + 2 * n + 1];
        float y  = (float)(h - 1 + TAP_I[n]) + oy;
        float xf = (float)(w - 1 + TAP_J[n]) + ox;
        y  = fminf(fmaxf(y, 0.0f), 63.0f);
        xf = fminf(fmaxf(xf, 0.0f), 63.0f);
        const float y0f = floorf(y);
        const float x0f = floorf(xf);
        const int y0 = (int)y0f;
        const int x0 = (int)x0f;
        const int y1 = (int)ceilf(y);
        const int x1 = (int)ceilf(xf);
        const int base = b * (HH * WW * CH);
        int4 o;
        o.x = base + (y0 * WW + x0) * CH;    // lt
        o.y = base + (y1 * WW + x0) * CH;    // rt
        o.z = base + (y0 * WW + x1) * CH;    // lb
        o.w = base + (y1 * WW + x1) * CH;    // rb
        sOfs[idx] = o;
        sFw[idx]  = make_float2(y - y0f, xf - x0f);
        sDst[idx] = p * A_STRIDE + n * CH;
    }
    __syncthreads();

    // ---- Stage 1: vectorized bilinear gather -> bf16 A panel -------------
    // Each lane owns 4 contiguous channels (16B-aligned since bases are
    // multiples of CH); each wave handles every 4th (pixel, tap) pair.
    {
        const int c4 = lane * 4;
        #pragma unroll 4
        for (int idx = wave; idx < NPAIR; idx += 4) {
            const int4   o  = sOfs[idx];     // wave-uniform broadcast
            const float2 fw = sFw[idx];
            const int    d  = sDst[idx];
            const float4 lt = *(const float4*)&xin[o.x + c4];  // global_load_b128
            const float4 rt = *(const float4*)&xin[o.y + c4];
            const float4 lb = *(const float4*)&xin[o.z + c4];
            const float4 rb = *(const float4*)&xin[o.w + c4];
            v4bf r;
            {
                const float vt = lt.x + (rt.x - lt.x) * fw.x;
                const float vb = lb.x + (rb.x - lb.x) * fw.x;
                r[0] = (__bf16)(vt + (vb - vt) * fw.y);
            }
            {
                const float vt = lt.y + (rt.y - lt.y) * fw.x;
                const float vb = lb.y + (rb.y - lb.y) * fw.x;
                r[1] = (__bf16)(vt + (vb - vt) * fw.y);
            }
            {
                const float vt = lt.z + (rt.z - lt.z) * fw.x;
                const float vb = lb.z + (rb.z - lb.z) * fw.x;
                r[2] = (__bf16)(vt + (vb - vt) * fw.y);
            }
            {
                const float vt = lt.w + (rt.w - lt.w) * fw.x;
                const float vb = lb.w + (rb.w - lb.w) * fw.x;
                r[3] = (__bf16)(vt + (vb - vt) * fw.y);
            }
            *(v4bf*)&sA[d + c4] = r;         // ds_store_b64
        }
    }
    __syncthreads();

    // ---- Stage 2: K-blocked WMMA GEMM (B fragments straight from L2) -----
    const int half = lane >> 4;          // K-half selector per ISA layouts
    const int lr   = lane & 15;          // A row / B column within tile
    const int col0 = (2 * wave)     * 16 + lr;   // wave owns 2 N-tiles
    const int col1 = (2 * wave + 1) * 16 + lr;

    v8f acc0 = {};
    v8f acc1 = {};

    #pragma unroll 6
    for (int ks = 0; ks < KSTEPS; ++ks) {
        const int kb = ks * 32;

        // A fragment (16-bit A 16x32 layout): lane half picks K groups
        const v8bf alo = *(const v8bf*)&sA[lr * A_STRIDE + kb + half * 16];
        const v8bf ahi = *(const v8bf*)&sA[lr * A_STRIDE + kb + half * 16 + 8];
        const v16bf a = __builtin_shufflevector(alo, ahi,
            0, 1, 2, 3, 4, 5, 6, 7, 8, 9, 10, 11, 12, 13, 14, 15);

        v16bf b0, b1;
        if (BF16W) {
            // lane = column, contiguous 16 K-values per half: 2x b128 each
            const v8bf b0lo = *(const v8bf*)&wt[(size_t)col0 * KTOT + kb + half * 16];
            const v8bf b0hi = *(const v8bf*)&wt[(size_t)col0 * KTOT + kb + half * 16 + 8];
            b0 = __builtin_shufflevector(b0lo, b0hi,
                0, 1, 2, 3, 4, 5, 6, 7, 8, 9, 10, 11, 12, 13, 14, 15);
            const v8bf b1lo = *(const v8bf*)&wt[(size_t)col1 * KTOT + kb + half * 16];
            const v8bf b1hi = *(const v8bf*)&wt[(size_t)col1 * KTOT + kb + half * 16 + 8];
            b1 = __builtin_shufflevector(b1lo, b1hi,
                0, 1, 2, 3, 4, 5, 6, 7, 8, 9, 10, 11, 12, 13, 14, 15);
        } else {
            // fallback: gather f32 weight column and convert
            #pragma unroll
            for (int kk = 0; kk < 16; ++kk) {
                const int krow = kb + half * 16 + kk;
                b0[kk] = (__bf16)wgt[(size_t)krow * FF + col0];
                b1[kk] = (__bf16)wgt[(size_t)krow * FF + col1];
            }
        }

        acc0 = __builtin_amdgcn_wmma_f32_16x16x32_bf16(
            false, a, false, b0, (short)0, acc0, false, false);
        acc1 = __builtin_amdgcn_wmma_f32_16x16x32_bf16(
            false, a, false, b1, (short)0, acc1, false, false);
    }

    // ---- Epilogue: bias + coalesced f32 stores ---------------------------
    const float bias0 = bias[col0];
    const float bias1 = bias[col1];
    #pragma unroll
    for (int v = 0; v < 8; ++v) {
        const int m = pbase + v + 8 * half;   // C/D layout: VGPR v, lane half
        out[(size_t)m * FF + col0] = acc0[v] + bias0;
        out[(size_t)m * FF + col1] = acc1[v] + bias1;
    }
}

// ---------------------------------------------------------------------------
extern "C" void kernel_launch(void* const* d_in, const int* in_sizes, int n_in,
                              void* d_out, int out_size, void* d_ws, size_t ws_size,
                              hipStream_t stream) {
    const float* x   = (const float*)d_in[0];
    const float* off = (const float*)d_in[1];
    const float* wgt = (const float*)d_in[2];
    const float* b   = (const float*)d_in[3];
    float* out = (float*)d_out;

    const int total_pixels = 8 * 64 * 64;      // B*H*W = 32768
    dim3 grid(total_pixels / 16);              // 2048 M-tiles
    dim3 block(128);                           // 4 waves (wave32)

    const size_t wt_bytes = (size_t)KTOT * FF * sizeof(__bf16);  // 294,912
    if (ws_size >= wt_bytes) {
        __bf16* wt = (__bf16*)d_ws;
        const int nelem = KTOT * FF;
        hipLaunchKernelGGL(weight_prepass_kernel, dim3((nelem + 255) / 256),
                           dim3(256), 0, stream, wgt, wt);
        hipLaunchKernelGGL((deform_conv_wmma_kernel<true>), grid, block, 0, stream,
                           x, off, wgt, wt, b, out);
    } else {
        hipLaunchKernelGGL((deform_conv_wmma_kernel<false>), grid, block, 0, stream,
                           x, off, wgt, (const __bf16*)nullptr, b, out);
    }
}
